// DIVEQ_85452669321693
// MI455X (gfx1250) — matrix operand order repaired and checked by reading, not compiled, not run on hardware
//
#include <hip/hip_runtime.h>
#include <math.h>

// ---------------------------------------------------------------------------
// VQ nearest-codebook search on gfx1250 (MI455X).
//   N=32768, D=256, K=8192 (N,K derived from in_sizes; D fixed at 256).
//   GEMM part (1.37e11 FLOP) via v_wmma_f32_16x16x32_bf16, bf16x3 split for
//   near-fp32 accuracy. Codebook (bf16 hi/lo, 8MB) stays L2-resident.
//   Codebook chunks staged with GLOBAL_LOAD_ASYNC_TO_LDS_B128 (ASYNCcnt) into
//   a double-buffered LDS tile so copies overlap the WMMA compute.
//   Two independent accumulator chains interleaved to avoid dependent-WMMA
//   issue stalls.
// Outputs (concatenated float): z_q [N*D], indices [N], perplexity [1].
// ---------------------------------------------------------------------------

typedef __attribute__((ext_vector_type(16))) __bf16 v16bf;
typedef __attribute__((ext_vector_type(8)))  float  v8f;

#define D_DIM 256
#define BM 128              // z rows per workgroup (16 per wave, 8 waves)
#define BN 32               // codebook rows per LDS chunk
#define LDS_STRIDE (D_DIM + 8)   // 264 bf16 -> 528B row stride, conflict-free
#define NOISE_VAR 0.001f

// --- codebook prep: split fp32 -> bf16 hi/lo, compute ||c||^2 per row -------
__global__ __launch_bounds__(256) void vq_prep_codebook(
    const float* __restrict__ cb, __bf16* __restrict__ cb_hi,
    __bf16* __restrict__ cb_lo, float* __restrict__ cnorm)
{
  const int row = blockIdx.x;
  const int t = threadIdx.x;                 // 0..255 == D
  const float f = cb[(size_t)row * D_DIM + t];
  const __bf16 h = (__bf16)f;
  cb_hi[(size_t)row * D_DIM + t] = h;
  cb_lo[(size_t)row * D_DIM + t] = (__bf16)(f - (float)h);

  float s = f * f;
  #pragma unroll
  for (int off = 16; off >= 1; off >>= 1) s += __shfl_xor(s, off, 32);
  __shared__ float wsum[8];
  if ((t & 31) == 0) wsum[t >> 5] = s;
  __syncthreads();
  if (t == 0) {
    float tot = 0.f;
    #pragma unroll
    for (int i = 0; i < 8; ++i) tot += wsum[i];
    cnorm[row] = tot;
  }
}

__global__ void vq_zero_counts(int* __restrict__ counts, int K) {
  int i = blockIdx.x * blockDim.x + threadIdx.x;
  if (i < K) counts[i] = 0;
}

// --- async global->LDS helpers (CDNA5 ASYNCcnt path) ------------------------
// Copies 64 contiguous bytes: LDS[dst+k] = MEM[src+k], 4 x b128 per lane.
__device__ __forceinline__ void async_copy_64B(unsigned lds_dst,
                                               const __bf16* gsrc) {
  unsigned long long g = (unsigned long long)(uintptr_t)gsrc;
  asm volatile(
      "global_load_async_to_lds_b128 %0, %1, off\n\t"
      "global_load_async_to_lds_b128 %0, %1, off offset:16\n\t"
      "global_load_async_to_lds_b128 %0, %1, off offset:32\n\t"
      "global_load_async_to_lds_b128 %0, %1, off offset:48"
      :: "v"(lds_dst), "v"(g) : "memory");
}

__device__ __forceinline__ void wait_asynccnt0() {
  asm volatile("s_wait_asynccnt 0" ::: "memory");
}

// --- main WMMA argmin kernel ------------------------------------------------
__global__ __launch_bounds__(256, 1) void vq_argmin(
    const float* __restrict__ z, const __bf16* __restrict__ cb_hi,
    const __bf16* __restrict__ cb_lo, const float* __restrict__ cnorm,
    int* __restrict__ out_idx, int K)
{
  // [buffer][hi/lo][row*stride+col]  => 2*2*32*264*2B = 67.6 KB
  __shared__ __bf16 ct[2][2][BN * LDS_STRIDE];

  const int tid  = threadIdx.x;
  const int wave = tid >> 5;
  const int lane = tid & 31;
  const int l = lane & 15;        // position within half-wave
  const int h = lane >> 4;        // half selector
  const int rowM = blockIdx.x * BM + wave * 16 + l;   // this lane's z row

  // Per-thread staging assignment for chunk copies (64B of hi + 64B of lo).
  const int lrow = tid >> 3;             // 0..31
  const int colB = (tid & 7) * 32;       // 0..224 elements, 64B-aligned
  const unsigned lds_h0 =
      (unsigned)(uintptr_t)&ct[0][0][lrow * LDS_STRIDE + colB];
  const unsigned lds_l0 =
      (unsigned)(uintptr_t)&ct[0][1][lrow * LDS_STRIDE + colB];
  const unsigned lds_h1 =
      (unsigned)(uintptr_t)&ct[1][0][lrow * LDS_STRIDE + colB];
  const unsigned lds_l1 =
      (unsigned)(uintptr_t)&ct[1][1][lrow * LDS_STRIDE + colB];

  // Load this wave's A fragments (16 z rows) into registers, split hi/lo.
  // 16-bit A 16x32 layout (matches B): element e -> K = 16*h + e
  v16bf a_hi[8], a_lo[8];
  {
    const float* zrow = z + (size_t)rowM * D_DIM;
    #pragma unroll
    for (int ks = 0; ks < 8; ++ks) {
      const int kBase = ks * 32;
      #pragma unroll
      for (int e = 0; e < 16; ++e) {
        const int kk = kBase + h * 16 + e;
        const float f = zrow[kk];
        const __bf16 hi = (__bf16)f;
        a_hi[ks][e] = hi;
        a_lo[ks][e] = (__bf16)(f - (float)hi);
      }
    }
  }

  float bestV[8];
  int   bestI[8];
  #pragma unroll
  for (int r = 0; r < 8; ++r) { bestV[r] = 3.4e38f; bestI[r] = 0; }

  const int nChunks = K / BN;

  // Prologue: kick off async copy of chunk 0 into buffer 0.
  {
    const size_t gofs = (size_t)lrow * D_DIM + colB;
    async_copy_64B(lds_h0, cb_hi + gofs);
    async_copy_64B(lds_l0, cb_lo + gofs);
  }

  // LDS row bases for this lane's two B subtiles (columns l and 16+l).
  const int tRow0 = (l)      * LDS_STRIDE;
  const int tRow1 = (16 + l) * LDS_STRIDE;

  for (int c = 0; c < nChunks; ++c) {
    const int buf = c & 1;
    // Chunk c has landed (our own async ops) ...
    wait_asynccnt0();
    // ... and everyone else's too.
    __syncthreads();

    // Prefetch chunk c+1 into the other buffer while we compute on this one.
    if (c + 1 < nChunks) {
      const size_t gofs = (size_t)((c + 1) * BN + lrow) * D_DIM + colB;
      if (buf == 0) {           // next goes into buffer 1
        async_copy_64B(lds_h1, cb_hi + gofs);
        async_copy_64B(lds_l1, cb_lo + gofs);
      } else {                  // next goes into buffer 0
        async_copy_64B(lds_h0, cb_hi + gofs);
        async_copy_64B(lds_l0, cb_lo + gofs);
      }
    }

    const int chunkBase = c * BN;
    const __bf16* lh = &ct[buf][0][0];
    const __bf16* ll = &ct[buf][1][0];

    // Two independent accumulator chains (subtile 0: codes +l, subtile 1:
    // codes +16+l) interleaved so consecutive WMMAs never depend on each
    // other -> no dependent-WMMA issue stalls, 6 WMMAs per ds_load group.
    v8f acc0 = {0.f, 0.f, 0.f, 0.f, 0.f, 0.f, 0.f, 0.f};
    v8f acc1 = {0.f, 0.f, 0.f, 0.f, 0.f, 0.f, 0.f, 0.f};
    #pragma unroll
    for (int ks = 0; ks < 8; ++ks) {
      const int kOfs = ks * 32 + h * 16;
      // 16-bit B 32x16 layout used here (matches A): lane = column, element e -> K = 16*h + e
      const __bf16* ph0 = &lh[tRow0 + kOfs];
      const __bf16* pl0 = &ll[tRow0 + kOfs];
      const __bf16* ph1 = &lh[tRow1 + kOfs];
      const __bf16* pl1 = &ll[tRow1 + kOfs];
      v16bf bh0, bl0, bh1, bl1;
      #pragma unroll
      for (int e = 0; e < 16; ++e) {
        bh0[e] = ph0[e]; bl0[e] = pl0[e];
        bh1[e] = ph1[e]; bl1[e] = pl1[e];
      }
      acc0 = __builtin_amdgcn_wmma_f32_16x16x32_bf16(
          false, a_hi[ks], false, bh0, (short)0, acc0, false, false);
      acc1 = __builtin_amdgcn_wmma_f32_16x16x32_bf16(
          false, a_hi[ks], false, bh1, (short)0, acc1, false, false);
      acc0 = __builtin_amdgcn_wmma_f32_16x16x32_bf16(
          false, a_hi[ks], false, bl0, (short)0, acc0, false, false);
      acc1 = __builtin_amdgcn_wmma_f32_16x16x32_bf16(
          false, a_hi[ks], false, bl1, (short)0, acc1, false, false);
      acc0 = __builtin_amdgcn_wmma_f32_16x16x32_bf16(
          false, a_lo[ks], false, bh0, (short)0, acc0, false, false);
      acc1 = __builtin_amdgcn_wmma_f32_16x16x32_bf16(
          false, a_lo[ks], false, bh1, (short)0, acc1, false, false);
    }

    // score(row r+8h, col) = ||c||^2 - 2 z.c  (||z||^2 constant per row).
    // Update subtile 0 first (smaller code indices) to keep first-occurrence
    // tie-break semantics of jnp.argmin.
    const int code0 = chunkBase + l;
    const int code1 = chunkBase + 16 + l;
    const float cn0 = cnorm[code0];             // L2-resident
    const float cn1 = cnorm[code1];
    #pragma unroll
    for (int r = 0; r < 8; ++r) {
      const float s0 = cn0 - 2.0f * acc0[r];
      if (s0 < bestV[r]) { bestV[r] = s0; bestI[r] = code0; }
      const float s1 = cn1 - 2.0f * acc1[r];
      if (s1 < bestV[r]) { bestV[r] = s1; bestI[r] = code1; }
    }
    // No trailing barrier: next iteration's wait+barrier protects both buffers.
  }

  // Reduce the 16 candidate lanes within each half; tie -> smaller index
  // (matches jnp.argmin first-occurrence since chunks scan ascending).
  #pragma unroll
  for (int r = 0; r < 8; ++r) {
    #pragma unroll
    for (int off = 8; off >= 1; off >>= 1) {
      const float ov = __shfl_xor(bestV[r], off, 32);
      const int   oi = __shfl_xor(bestI[r], off, 32);
      if (ov < bestV[r] || (ov == bestV[r] && oi < bestI[r])) {
        bestV[r] = ov; bestI[r] = oi;
      }
    }
    if (l == 0)
      out_idx[blockIdx.x * BM + wave * 16 + r + 8 * h] = bestI[r];
  }
}

// --- gather + directional-noise quantization + histogram -------------------
__global__ __launch_bounds__(256) void vq_quantize(
    const float* __restrict__ z, const float* __restrict__ cb,
    const float* __restrict__ noise, const int* __restrict__ idx,
    float* __restrict__ zq, float* __restrict__ oidx, int* __restrict__ counts)
{
  const int n = blockIdx.x;
  const int t = threadIdx.x;                  // 0..255 == D
  const int k = idx[n];
  const float zv  = z[(size_t)n * D_DIM + t];
  const float cv  = cb[(size_t)k * D_DIM + t];
  const float dir = cv - zv;
  const float rv  = noise[(size_t)n * D_DIM + t] * NOISE_VAR + dir;

  float s1 = rv * rv, s2 = dir * dir;
  #pragma unroll
  for (int off = 16; off >= 1; off >>= 1) {
    s1 += __shfl_xor(s1, off, 32);
    s2 += __shfl_xor(s2, off, 32);
  }
  __shared__ float a1[8], a2[8];
  if ((t & 31) == 0) { a1[t >> 5] = s1; a2[t >> 5] = s2; }
  __syncthreads();
  float t1 = 0.f, t2 = 0.f;
  #pragma unroll
  for (int i = 0; i < 8; ++i) { t1 += a1[i]; t2 += a2[i]; }

  const float inv = 1.0f / fmaxf(sqrtf(t1), 1e-12f);  // 1/max(||rv||,eps)
  const float em  = sqrtf(t2);                         // ||direction||
  zq[(size_t)n * D_DIM + t] = zv + em * rv * inv;
  if (t == 0) { oidx[n] = (float)k; atomicAdd(&counts[k], 1); }
}

// --- perplexity -------------------------------------------------------------
__global__ __launch_bounds__(256) void vq_perplexity(
    const int* __restrict__ counts, float* __restrict__ out, int K, float invN)
{
  float s = 0.f;
  for (int k = threadIdx.x; k < K; k += 256) {
    const float p = (float)counts[k] * invN;
    s += p * logf(p + 1e-10f);
  }
  #pragma unroll
  for (int off = 16; off >= 1; off >>= 1) s += __shfl_xor(s, off, 32);
  __shared__ float wsum[8];
  if ((threadIdx.x & 31) == 0) wsum[threadIdx.x >> 5] = s;
  __syncthreads();
  if (threadIdx.x == 0) {
    float tot = 0.f;
    #pragma unroll
    for (int i = 0; i < 8; ++i) tot += wsum[i];
    out[0] = expf(-tot);
  }
}

// ---------------------------------------------------------------------------
extern "C" void kernel_launch(void* const* d_in, const int* in_sizes, int n_in,
                              void* d_out, int out_size, void* d_ws, size_t ws_size,
                              hipStream_t stream) {
  const float* z  = (const float*)d_in[0];
  const float* cb = (const float*)d_in[1];
  const float* nz = (const float*)d_in[2];
  const int N = in_sizes[0] / D_DIM;
  const int K = in_sizes[1] / D_DIM;

  // Workspace layout (~8.6 MB total)
  char* ws = (char*)d_ws;
  __bf16* cb_hi = (__bf16*)ws;  ws += (size_t)K * D_DIM * 2;
  __bf16* cb_lo = (__bf16*)ws;  ws += (size_t)K * D_DIM * 2;
  float*  cnorm = (float*)ws;   ws += (size_t)K * 4;
  int*    idx   = (int*)ws;     ws += (size_t)N * 4;
  int*    counts= (int*)ws;

  float* zq = (float*)d_out;
  float* oi = zq + (size_t)N * D_DIM;
  float* op = oi + N;

  vq_zero_counts  <<<(K + 255) / 256, 256, 0, stream>>>(counts, K);
  vq_prep_codebook<<<K,              256, 0, stream>>>(cb, cb_hi, cb_lo, cnorm);
  vq_argmin       <<<N / BM,         256, 0, stream>>>(z, cb_hi, cb_lo, cnorm, idx, K);
  vq_quantize     <<<N,              256, 0, stream>>>(z, cb, nz, idx, zq, oi, counts);
  vq_perplexity   <<<1,              256, 0, stream>>>(counts, op, K, 1.0f / (float)N);
}
